// Model_9586367004699
// MI455X (gfx1250) — compile-verified, matrix operand order checked by reading, not run on hardware
//
#include <hip/hip_runtime.h>
#include <hip/hip_bf16.h>
#include <math.h>

// ---------------------------------------------------------------------------
// Problem constants (from reference): B=8, N=2048, DM=1024, H=16, D=32, FF=4096
// ---------------------------------------------------------------------------
#define MTOK   16384          // B*N tokens
#define DMODEL 1024
#define FFDIM  4096
#define NHEAD  16
#define HDIM   32             // per-head dim for q/k (v heads are 2*HDIM)
#define LAMBDA_INIT 0.35550907f          // 0.8 - 0.6*exp(-0.3)
#define ONE_MINUS_LAMBDA 0.64449093f

typedef __attribute__((ext_vector_type(16))) _Float16 v16h;
typedef __attribute__((ext_vector_type(8)))  float    v8f;
typedef __attribute__((ext_vector_type(8)))  unsigned int v8u;
typedef __attribute__((ext_vector_type(4)))  int      v4i;

union FragU { v8u u; v16h h; };

#if defined(__HIP_DEVICE_COMPILE__) && __has_builtin(__builtin_amdgcn_global_load_async_to_lds_b128)
#define USE_ASYNC_LDS 1
typedef __attribute__((address_space(1))) v4i GInt4;   // global int4
typedef __attribute__((address_space(3))) v4i LInt4;   // LDS int4
#else
#define USE_ASYNC_LDS 0
#endif

__device__ __forceinline__ void wait_async_zero() {
#if USE_ASYNC_LDS
#if __has_builtin(__builtin_amdgcn_s_wait_asynccnt)
  __builtin_amdgcn_s_wait_asynccnt(0);
#else
  asm volatile("s_wait_asynccnt 0x0" ::: "memory");
#endif
#endif
}

// ---------------------------------------------------------------------------
// fp32 -> fp16 conversion (weights)
// ---------------------------------------------------------------------------
__global__ __launch_bounds__(256) void cvt_f32_to_f16(const float* __restrict__ in,
                                                      _Float16* __restrict__ out, int n) {
  for (int i = blockIdx.x * 256 + threadIdx.x; i < n; i += gridDim.x * 256)
    out[i] = (_Float16)in[i];
}

// ---------------------------------------------------------------------------
// LayerNorm over last dim (1024), fp32 in -> fp16 out. One block per row.
// ---------------------------------------------------------------------------
__global__ __launch_bounds__(256) void layernorm_f16(const float* __restrict__ x,
                                                     const float* __restrict__ g,
                                                     const float* __restrict__ b,
                                                     _Float16* __restrict__ out) {
  const int row = blockIdx.x;
  const int t   = threadIdx.x;
  const float* xr = x + (size_t)row * DMODEL;

  float v[4], s = 0.f, ss = 0.f;
#pragma unroll
  for (int i = 0; i < 4; ++i) {
    v[i] = xr[t + 256 * i];
    s  += v[i];
    ss += v[i] * v[i];
  }
  __shared__ float r1[256], r2[256];
  r1[t] = s; r2[t] = ss;
  __syncthreads();
  for (int off = 128; off > 0; off >>= 1) {
    if (t < off) { r1[t] += r1[t + off]; r2[t] += r2[t + off]; }
    __syncthreads();
  }
  const float mean = r1[0] * (1.0f / DMODEL);
  const float var  = r2[0] * (1.0f / DMODEL) - mean * mean;
  const float inv  = rsqrtf(var + 1e-5f);
#pragma unroll
  for (int i = 0; i < 4; ++i) {
    const int c = t + 256 * i;
    out[(size_t)row * DMODEL + c] = (_Float16)((v[i] - mean) * inv * g[c] + b[c]);
  }
}

// ---------------------------------------------------------------------------
// WMMA GEMM:  out[M,N] = A[M,K] @ W[N,K]^T  (+bias)(+residual f32)(relu?)
// Block tile 128x128, BK=32, double-buffered LDS, 8 waves; each wave owns a
// 32x64 patch = 2x4 v_wmma_f32_16x16x32_f16 tiles (8 WMMA / wave / K-chunk).
// Tiles staged with GLOBAL_LOAD_ASYNC_TO_LDS_B128 when available (ASYNCcnt),
// single workgroup barrier per K-chunk.
// ---------------------------------------------------------------------------
__device__ __forceinline__ void stage_tile_128x32(
    const _Float16* __restrict__ gbase,   // matrix base
    int firstRow, int K, int k0,
    _Float16* __restrict__ lbuf,          // 128*32 halves
    int tid) {
#pragma unroll
  for (int rep = 0; rep < 2; ++rep) {
    const int u = tid + 256 * rep;        // uint4 unit: 4 per row
    const int r = u >> 2;
    const int q = u & 3;
    const _Float16* gp = gbase + (size_t)(firstRow + r) * K + k0 + q * 8;
#if USE_ASYNC_LDS
    __builtin_amdgcn_global_load_async_to_lds_b128(
        (GInt4*)(void*)gp, (LInt4*)(void*)(lbuf + u * 8), 0, 0);
#else
    ((uint4*)lbuf)[u] = *(const uint4*)gp;
#endif
  }
}

__global__ __launch_bounds__(256) void wmma_gemm_f16(
    const _Float16* __restrict__ A, const _Float16* __restrict__ W,
    const float* __restrict__ bias, const float* __restrict__ res,
    float* __restrict__ outF, _Float16* __restrict__ outH,
    int M, int N, int K, int relu) {
  __shared__ _Float16 sA[2][128 * 32];
  __shared__ _Float16 sB[2][128 * 32];

  const int tid    = threadIdx.x;
  const int waveId = tid >> 5;
  const int lane   = tid & 31;
  const int lhalf  = lane & 15;   // column / row-within-16
  const int lhi    = lane >> 4;   // half-wave group

  const int wm = waveId >> 1;     // 0..3 : 32-row strip
  const int wn = waveId & 1;      // 0..1 : 64-col strip

  const int rowBase = blockIdx.y * 128;
  const int colBase = blockIdx.x * 128;

  v8f acc[2][4];
#pragma unroll
  for (int i = 0; i < 2; ++i)
#pragma unroll
    for (int j = 0; j < 4; ++j) acc[i][j] = (v8f){0.f,0.f,0.f,0.f,0.f,0.f,0.f,0.f};

  const int nCh = K >> 5;

  // prologue: chunk 0 -> buffer 0
  stage_tile_128x32(A, rowBase, K, 0, sA[0], tid);
  stage_tile_128x32(W, colBase, K, 0, sB[0], tid);

  for (int c = 0; c < nCh; ++c) {
    const int p = c & 1;

    // own async loads for chunk c complete; barrier publishes all waves'
    // LDS writes AND guards the buffer we are about to overwrite (WAR).
    wait_async_zero();
    __syncthreads();

    if (c + 1 < nCh) {
      stage_tile_128x32(A, rowBase, K, (c + 1) << 5, sA[1 - p], tid);
      stage_tile_128x32(W, colBase, K, (c + 1) << 5, sB[1 - p], tid);
    }

    const unsigned int* sAd = (const unsigned int*)sA[p];
    const unsigned int* sBd = (const unsigned int*)sB[p];

    // fragments: lane<16 holds K pairs {0..7,16..23}, lane>=16 {8..15,24..31}
    FragU fa[2], fb[4];
#pragma unroll
    for (int i = 0; i < 2; ++i) {
      const int r = wm * 32 + i * 16 + lhalf;
#pragma unroll
      for (int d = 0; d < 8; ++d) {
        const int kh = ((d & 4) << 2) + ((d & 3) << 1) + (lhi << 3);
        fa[i].u[d] = sAd[(r * 32 + kh) >> 1];
      }
    }
#pragma unroll
    for (int j = 0; j < 4; ++j) {
      const int r = wn * 64 + j * 16 + lhalf;
#pragma unroll
      for (int d = 0; d < 8; ++d) {
        const int kh = ((d & 4) << 2) + ((d & 3) << 1) + (lhi << 3);
        fb[j].u[d] = sBd[(r * 32 + kh) >> 1];
      }
    }

#pragma unroll
    for (int i = 0; i < 2; ++i)
#pragma unroll
      for (int j = 0; j < 4; ++j)
        acc[i][j] = __builtin_amdgcn_wmma_f32_16x16x32_f16(
            false, fa[i].h, false, fb[j].h, (short)0, acc[i][j], false, false);
  }

  // ---- epilogue ----
#pragma unroll
  for (int i = 0; i < 2; ++i) {
#pragma unroll
    for (int j = 0; j < 4; ++j) {
      const int gcol = colBase + wn * 64 + j * 16 + lhalf;
#pragma unroll
      for (int r = 0; r < 8; ++r) {
        const int grow = rowBase + wm * 32 + i * 16 + r + 8 * lhi;
        float o = acc[i][j][r];
        if (bias) o += bias[gcol];
        if (res)  o += res[(size_t)grow * N + gcol];
        if (relu) o = fmaxf(o, 0.f);
        if (outF) outF[(size_t)grow * N + gcol] = o;
        if (outH) outH[(size_t)grow * N + gcol] = (_Float16)o;
      }
    }
  }
}

// ---------------------------------------------------------------------------
// Per-token differential attention. One wave (32 lanes) per token; 8 waves/blk.
// q,k,v are the f16 GEMM outputs [tok, 1024]. Output o16 [tok, 1024] f16.
// ---------------------------------------------------------------------------
__global__ __launch_bounds__(256) void diff_attn_kernel(
    const _Float16* __restrict__ q16, const _Float16* __restrict__ k16,
    const _Float16* __restrict__ v16,
    const float* __restrict__ lq1, const float* __restrict__ lq2,
    const float* __restrict__ lk1, const float* __restrict__ lk2,
    _Float16* __restrict__ o16) {
  __shared__ _Float16 sk[8][32 * 32];    // k' rows: [2H=32][D=32]
  __shared__ _Float16 sv[8][16 * 64];    // v' rows: [H=16][2D=64]
  __shared__ float    satt[8][16 * 16];  // att2 weights

  const int waveId = threadIdx.x >> 5;
  const int lane   = threadIdx.x & 31;
  const int tok    = blockIdx.x * 8 + waveId;

  const float scale = 0.17677669529663687f;   // 32^-0.5
  const int j = lane >> 4;      // q/k group 0|1
  const int h = lane & 15;      // head within group

  // per-lane lambda (cheap: 2x 32-length dots)
  float d1 = 0.f, d2 = 0.f;
#pragma unroll 8
  for (int i = 0; i < HDIM; ++i) { d1 += lq1[i] * lk1[i]; d2 += lq2[i] * lk2[i]; }
  const float lam = __expf(d1) - __expf(d2) + LAMBDA_INIT;

  const size_t base = (size_t)tok * DMODEL;

  // q' row for this lane (head = j*16+h), with scale
  float qrow[HDIM];
#pragma unroll
  for (int d = 0; d < HDIM; ++d)
    qrow[d] = (float)q16[base + h * 64 + d * 2 + j] * scale;

  // stage k' (lane writes its own row) and v' (straight copy) to LDS
#pragma unroll
  for (int d = 0; d < HDIM; ++d)
    sk[waveId][lane * 32 + d] = k16[base + h * 64 + d * 2 + j];
#pragma unroll
  for (int c = 0; c < 32; ++c)
    sv[waveId][lane * 32 + c] = v16[base + lane * 32 + c];
  __syncthreads();

  // att row for this lane: 32 scores over d=32, then softmax over 32
  float att[32];
#pragma unroll
  for (int kk = 0; kk < 32; ++kk) {
    float s = 0.f;
#pragma unroll
    for (int d = 0; d < HDIM; ++d)
      s += qrow[d] * (float)sk[waveId][kk * 32 + d];
    att[kk] = s;
  }
  float m = att[0];
#pragma unroll
  for (int kk = 1; kk < 32; ++kk) m = fmaxf(m, att[kk]);
  float ssum = 0.f;
#pragma unroll
  for (int kk = 0; kk < 32; ++kk) { att[kk] = __expf(att[kk] - m); ssum += att[kk]; }
  const float rinv = 1.f / ssum;
#pragma unroll
  for (int kk = 0; kk < 32; ++kk) att[kk] *= rinv;

  // att2[a,hk] = att[a,hk] - lam * att[16+a,16+hk]  (valid in lanes 0..15)
#pragma unroll
  for (int hk = 0; hk < 16; ++hk) {
    const float other = __shfl_down(att[16 + hk], 16);
    const float w = att[hk] - lam * other;
    if (lane < 16) satt[waveId][lane * 16 + hk] = w;
  }
  __syncthreads();

  // out[a, e] = sum_h att2[a,h] * v'[h,e]; lanes split e in halves of 32
  const int a  = lane & 15;
  const int eh = lane >> 4;
  float outv[32];
  float ssq = 0.f;
#pragma unroll
  for (int e = 0; e < 32; ++e) {
    const int eg = eh * 32 + e;
    float s = 0.f;
#pragma unroll
    for (int hk = 0; hk < 16; ++hk)
      s += satt[waveId][a * 16 + hk] * (float)sv[waveId][hk * 64 + eg];
    outv[e] = s;
    ssq += s * s;
  }
  const float tot = ssq + __shfl_xor(ssq, 16);
  const float rms = rsqrtf(tot * (1.0f / 64.0f) + 1e-5f) * ONE_MINUS_LAMBDA;
#pragma unroll
  for (int e = 0; e < 32; ++e) {
    const int eg = eh * 32 + e;
    o16[base + a * 64 + eg] = (_Float16)(outv[e] * rms);
  }
}

// ---------------------------------------------------------------------------
// launcher
// ---------------------------------------------------------------------------
extern "C" void kernel_launch(void* const* d_in, const int* in_sizes, int n_in,
                              void* d_out, int out_size, void* d_ws, size_t ws_size,
                              hipStream_t stream) {
  const float* x    = (const float*)d_in[0];
  const float* Wq   = (const float*)d_in[1];
  const float* Wk   = (const float*)d_in[2];
  const float* Wv   = (const float*)d_in[3];
  const float* lq1  = (const float*)d_in[4];
  const float* lq2  = (const float*)d_in[5];
  const float* lk1  = (const float*)d_in[6];
  const float* lk2  = (const float*)d_in[7];
  const float* Wo   = (const float*)d_in[8];
  const float* bo   = (const float*)d_in[9];
  const float* ln1g = (const float*)d_in[10];
  const float* ln1b = (const float*)d_in[11];
  const float* ln2g = (const float*)d_in[12];
  const float* ln2b = (const float*)d_in[13];
  const float* W1   = (const float*)d_in[14];
  const float* b1   = (const float*)d_in[15];
  const float* W2   = (const float*)d_in[16];
  const float* b2   = (const float*)d_in[17];
  float* out = (float*)d_out;

  // workspace carve-up (bytes); S = one [16384,1024] f16 activation
  const size_t S = (size_t)MTOK * DMODEL * sizeof(_Float16);  // 32 MiB
  char* ws = (char*)d_ws;
  _Float16* h16   = (_Float16*)(ws + 0 * S);
  _Float16* q16   = (_Float16*)(ws + 1 * S);
  _Float16* k16   = (_Float16*)(ws + 2 * S);
  _Float16* v16   = (_Float16*)(ws + 3 * S);
  _Float16* o16   = (_Float16*)(ws + 4 * S);
  _Float16* h2_16 = (_Float16*)(ws + 5 * S);
  float*    conn  = (float*)   (ws + 6 * S);            // f32, 2*S bytes
  _Float16* a1_16 = (_Float16*)(ws + 8 * S);            // [16384,4096] f16 = 4*S
  char* wbase = ws + 12 * S;
  _Float16* Wq16 = (_Float16*)(wbase);
  _Float16* Wk16 = (_Float16*)(wbase + 1 * (size_t)DMODEL * DMODEL * 2);
  _Float16* Wv16 = (_Float16*)(wbase + 2 * (size_t)DMODEL * DMODEL * 2);
  _Float16* Wo16 = (_Float16*)(wbase + 3 * (size_t)DMODEL * DMODEL * 2);
  _Float16* W116 = (_Float16*)(wbase + 4 * (size_t)DMODEL * DMODEL * 2);
  _Float16* W216 = (_Float16*)(wbase + 4 * (size_t)DMODEL * DMODEL * 2
                                     + (size_t)FFDIM * DMODEL * 2);

  const int nW  = DMODEL * DMODEL;   // 1M
  const int nWf = FFDIM * DMODEL;    // 4M
  cvt_f32_to_f16<<<2048, 256, 0, stream>>>(Wq, Wq16, nW);
  cvt_f32_to_f16<<<2048, 256, 0, stream>>>(Wk, Wk16, nW);
  cvt_f32_to_f16<<<2048, 256, 0, stream>>>(Wv, Wv16, nW);
  cvt_f32_to_f16<<<2048, 256, 0, stream>>>(Wo, Wo16, nW);
  cvt_f32_to_f16<<<4096, 256, 0, stream>>>(W1, W116, nWf);
  cvt_f32_to_f16<<<4096, 256, 0, stream>>>(W2, W216, nWf);

  // LN1
  layernorm_f16<<<MTOK, 256, 0, stream>>>(x, ln1g, ln1b, h16);

  // QKV projections
  dim3 g1(DMODEL / 128, MTOK / 128);
  wmma_gemm_f16<<<g1, 256, 0, stream>>>(h16, Wq16, nullptr, nullptr, nullptr, q16,
                                        MTOK, DMODEL, DMODEL, 0);
  wmma_gemm_f16<<<g1, 256, 0, stream>>>(h16, Wk16, nullptr, nullptr, nullptr, k16,
                                        MTOK, DMODEL, DMODEL, 0);
  wmma_gemm_f16<<<g1, 256, 0, stream>>>(h16, Wv16, nullptr, nullptr, nullptr, v16,
                                        MTOK, DMODEL, DMODEL, 0);

  // per-token differential attention
  diff_attn_kernel<<<MTOK / 8, 256, 0, stream>>>(q16, k16, v16, lq1, lq2, lk1, lk2, o16);

  // Wo projection + bias + residual(x) -> conn (f32)
  wmma_gemm_f16<<<g1, 256, 0, stream>>>(o16, Wo16, bo, x, conn, nullptr,
                                        MTOK, DMODEL, DMODEL, 0);

  // LN2
  layernorm_f16<<<MTOK, 256, 0, stream>>>(conn, ln2g, ln2b, h2_16);

  // FFN1: relu(h2 @ W1^T + b1) -> a1 (f16)
  dim3 g2(FFDIM / 128, MTOK / 128);
  wmma_gemm_f16<<<g2, 256, 0, stream>>>(h2_16, W116, b1, nullptr, nullptr, a1_16,
                                        MTOK, FFDIM, DMODEL, 1);

  // FFN2: a1 @ W2^T + b2 + conn -> out (f32)
  wmma_gemm_f16<<<g1, 256, 0, stream>>>(a1_16, W216, b2, conn, out, nullptr,
                                        MTOK, DMODEL, FFDIM, 0);
}
